// FastEquiformerV2StressHead_1949915152407
// MI455X (gfx1250) — matrix operand order repaired and checked by reading, not compiled, not run on hardware
//
#include <hip/hip_runtime.h>
#include <hip/hip_bf16.h>

typedef __attribute__((ext_vector_type(16))) _Float16 v16h;
typedef __attribute__((ext_vector_type(8)))  float    v8f;

__device__ __forceinline__ v8f wmma_f16(v16h a, v16h b, v8f c) {
  // D = A(16x32 f16) x B(32x16 f16) + C(16x16 f32)
  return __builtin_amdgcn_wmma_f32_16x16x32_f16(false, a, false, b, (short)0, c, false, false);
}

__device__ __forceinline__ void wait_ds() {
  asm volatile("s_wait_dscnt 0x0" ::: "memory");
}

// order-preserving float<->uint for atomicMax on signed floats
__device__ __forceinline__ unsigned ord_enc(float f) {
  unsigned u = __float_as_uint(f);
  return (u & 0x80000000u) ? ~u : (u | 0x80000000u);
}
__device__ __forceinline__ float ord_dec(unsigned u) {
  return (u & 0x80000000u) ? __uint_as_float(u & 0x7fffffffu) : __uint_as_float(~u);
}

__device__ __forceinline__ float silu_f(float x) { return x / (1.0f + __expf(-x)); }
__device__ __forceinline__ float lrelu_f(float x) { return x > 0.0f ? x : 0.2f * x; }

// A fragment (16x32 f16) from global f32 row (row-major, 128 ch). A layout used here (contiguous k, same order as B):
// lane m=lane&15; lanes<16 hold K={0..15}+kt*32, lanes>=16 hold K={16..31}+kt*32,
// packed two contiguous halves per VGPR.
__device__ __forceinline__ v16h load_a_global(const float* row, int kt, int lane) {
  const int kbase = (lane & 16) ? 16 : 0;
  v16h a;
#pragma unroll
  for (int v = 0; v < 8; ++v) {
    const int K = kt * 32 + kbase + v * 2;
    const float2 p = *(const float2*)(row + K);
    a[2 * v]     = (_Float16)p.x;
    a[2 * v + 1] = (_Float16)p.y;
  }
  return a;
}

// A fragment (16x32) from an LDS f16 row of 32 halves (K pairs contiguous)
__device__ __forceinline__ v16h load_a_lds32(const _Float16* row, int lane) {
  const int kbase = (lane & 16) ? 16 : 0;
  v16h a;
#pragma unroll
  for (int v = 0; v < 8; ++v) {
    const int K = kbase + v * 2;
    a[2 * v]     = row[K];
    a[2 * v + 1] = row[K + 1];
  }
  return a;
}

// B fragment: lane holds col N=lane&15 (+ntile*16), K halves contiguous (16 of them, 32B)
__device__ __forceinline__ v16h load_b_lds(const _Float16* p) {
  return *(const v16h*)p;
}

// ---------------- kernel 0: init accumulators ----------------
__global__ void k_init(unsigned* nodemax, float* denom, float* node_out, float* out,
                       int N, long long total_no, int out_n) {
  long long i = (long long)blockIdx.x * blockDim.x + threadIdx.x;
  long long stride = (long long)gridDim.x * blockDim.x;
  for (long long t = i; t < total_no; t += stride) node_out[t] = 0.0f;
  for (long long t = i; t < (long long)N * 8; t += stride) { nodemax[t] = 0u; denom[t] = 0.0f; }
  for (long long t = i; t < out_n; t += stride) out[t] = 0.0f;
}

// ---------------- kernel 1: radial features rad[E][3] ----------------
__global__ void __launch_bounds__(256)
k_rad(const float* edge_distance, const int* atomic_numbers, const int* edge_index,
      const float* elem_emb, const float* W_rbf, const float* W_rad,
      float* rad, int E, int maxz) {
  __shared__ float s_emb[90 * 32];
  __shared__ float s_wrbf[32 * 32];
  __shared__ float s_wrad[32 * 3];
  for (int t = threadIdx.x; t < maxz * 32; t += blockDim.x) s_emb[t] = elem_emb[t];
  for (int t = threadIdx.x; t < 32 * 32; t += blockDim.x) s_wrbf[t] = W_rbf[t];
  for (int t = threadIdx.x; t < 32 * 3; t += blockDim.x) s_wrad[t] = W_rad[t];
  __syncthreads();
  int i = blockIdx.x * blockDim.x + threadIdx.x;
  int stride = gridDim.x * blockDim.x;
  const float step = 8.0f / 31.0f;
  for (int e = i; e < E; e += stride) {
    float d = edge_distance[e];
    float rbf[32];
#pragma unroll
    for (int j = 0; j < 32; ++j) {
      float x = (d - step * (float)j) * 2.0f;
      rbf[j] = __expf(-x * x);
    }
    int zs = atomic_numbers[edge_index[e]];
    int zd = atomic_numbers[edge_index[E + e]];
    float r0 = 0.0f, r1 = 0.0f, r2 = 0.0f;
    for (int k = 0; k < 32; ++k) {
      float f = s_emb[zs * 32 + k] + s_emb[zd * 32 + k];
#pragma unroll
      for (int j = 0; j < 32; ++j) f += rbf[j] * s_wrbf[j * 32 + k];
      f = silu_f(f);
      r0 += f * s_wrad[k * 3 + 0];
      r1 += f * s_wrad[k * 3 + 1];
      r2 += f * s_wrad[k * 3 + 2];
    }
    rad[(size_t)e * 3 + 0] = r0;
    rad[(size_t)e * 3 + 1] = r1;
    rad[(size_t)e * 3 + 2] = r2;
  }
}

// ---------------- kernel 2: logits + segment max (WMMA) ----------------
__global__ void __launch_bounds__(256)
k_logits(const float* node_emb, const int* edge_index,
         const float* W_src, const float* W_dst, const float* W_alpha,
         const float* alpha_vec, const float* rad,
         float* logits, unsigned* nodemax, int E) {
  __shared__ __align__(32) _Float16 WsT[32][128];   // [h][c] = W_src[0][c][h]
  __shared__ __align__(32) _Float16 WdT[32][128];
  __shared__ __align__(32) _Float16 WaT[128][32];   // [o][h] = W_alpha[h][o]
  __shared__ float s_avec[128];
  __shared__ int   s_eidx[8][16];
  __shared__ int   s_didx[8][16];
  __shared__ float s_rad0[8][16];
  __shared__ __align__(32) _Float16 s_m0[8][16][32];

  for (int t = threadIdx.x; t < 32 * 128; t += blockDim.x) {
    int o = t >> 7, c = t & 127;
    WsT[o][c] = (_Float16)W_src[c * 32 + o];
    WdT[o][c] = (_Float16)W_dst[c * 32 + o];
  }
  for (int t = threadIdx.x; t < 128 * 32; t += blockDim.x) {
    int o = t >> 5, h = t & 31;
    WaT[o][h] = (_Float16)W_alpha[h * 128 + o];
  }
  for (int t = threadIdx.x; t < 128; t += blockDim.x) s_avec[t] = alpha_vec[t];
  __syncthreads();

  const int lane  = threadIdx.x & 31;
  const int wid   = threadIdx.x >> 5;
  const int gwave = blockIdx.x * 8 + wid;
  const int nwaves = gridDim.x * 8;
  const int ntiles = (E + 15) >> 4;
  const int n     = lane & 15;
  const int rbase = (lane & 16) ? 8 : 0;
  const int koff  = (lane & 16) ? 16 : 0;

  for (int tile = gwave; tile < ntiles; tile += nwaves) {
    const int e0 = tile << 4;
    if (lane < 16) {
      int e = e0 + lane;
      int ec = e < E ? e : E - 1;
      s_eidx[wid][lane] = edge_index[ec];
      s_didx[wid][lane] = edge_index[E + ec];
      s_rad0[wid][lane] = rad[(size_t)ec * 3];
    }
    wait_ds();
    const float* srow = node_emb + (size_t)s_eidx[wid][n] * (9 * 128);
    const float* drow = node_emb + (size_t)s_didx[wid][n] * (9 * 128);
    v8f acc0 = {}, acc1 = {};
#pragma unroll
    for (int kt = 0; kt < 4; ++kt) {
      v16h aS = load_a_global(srow, kt, lane);
      acc0 = wmma_f16(aS, load_b_lds(&WsT[n][kt * 32 + koff]), acc0);
      acc1 = wmma_f16(aS, load_b_lds(&WsT[16 + n][kt * 32 + koff]), acc1);
      v16h aD = load_a_global(drow, kt, lane);
      acc0 = wmma_f16(aD, load_b_lds(&WdT[n][kt * 32 + koff]), acc0);
      acc1 = wmma_f16(aD, load_b_lds(&WdT[16 + n][kt * 32 + koff]), acc1);
    }
    // scale by rad0, stash as f16 tile for the transposed second GEMM
#pragma unroll
    for (int r = 0; r < 8; ++r) {
      float s = s_rad0[wid][rbase + r];
      s_m0[wid][rbase + r][n]      = (_Float16)(acc0[r] * s);
      s_m0[wid][rbase + r][16 + n] = (_Float16)(acc1[r] * s);
    }
    wait_ds();
    v16h aM = load_a_lds32(&s_m0[wid][n][0], lane);
#pragma unroll
    for (int head = 0; head < 8; ++head) {
      v8f c = {};
      c = wmma_f16(aM, load_b_lds(&WaT[head * 16 + n][koff]), c);
      float av = s_avec[head * 16 + n];
#pragma unroll
      for (int r = 0; r < 8; ++r) {
        float t = lrelu_f(c[r]) * av;
        t += __shfl_xor(t, 1, 32);
        t += __shfl_xor(t, 2, 32);
        t += __shfl_xor(t, 4, 32);
        t += __shfl_xor(t, 8, 32);
        if (n == 0) {
          int m = rbase + r;
          int e = e0 + m;
          if (e < E) {
            logits[(size_t)e * 8 + head] = t;
            atomicMax(&nodemax[(size_t)s_didx[wid][m] * 8 + head], ord_enc(t));
          }
        }
      }
    }
  }
}

// ---------------- kernel 3: softmax denominator ----------------
__global__ void k_denom(const float* logits, const int* edge_index,
                        const unsigned* nodemax, float* denom, int E) {
  long long i = (long long)blockIdx.x * blockDim.x + threadIdx.x;
  long long total = (long long)E * 8;
  long long stride = (long long)gridDim.x * blockDim.x;
  for (long long t = i; t < total; t += stride) {
    int e = (int)(t >> 3), h = (int)(t & 7);
    int d = edge_index[E + e];
    float mx = ord_dec(nodemax[(size_t)d * 8 + h]);
    atomicAdd(&denom[(size_t)d * 8 + h], __expf(logits[t] - mx));
  }
}

// ---------------- kernel 4: msg + value + attention scatter (WMMA) ----------------
__global__ void __launch_bounds__(128)
k_msgval(const float* node_emb, const int* edge_index,
         const float* W_src, const float* W_dst, const float* W_val,
         const float* rad, const float* logits, const unsigned* nodemax,
         const float* denom, float* node_out, int E) {
  __shared__ __align__(32) _Float16 WsT[3][32][128];  // [d][h][c]
  __shared__ __align__(32) _Float16 WdT[3][32][128];
  __shared__ __align__(32) _Float16 WvT[3][32][32];   // [d][o][h]
  __shared__ int   s_eidx[4][16];
  __shared__ int   s_didx[4][16];
  __shared__ float s_rad[4][16][3];
  __shared__ float s_alpha[4][16][8];
  __shared__ __align__(32) _Float16 s_ms[4][16][32];

  for (int t = threadIdx.x; t < 3 * 32 * 128; t += blockDim.x) {
    int d = t >> 12, rem = t & 4095;
    int o = rem >> 7, c = rem & 127;
    WsT[d][o][c] = (_Float16)W_src[((size_t)d * 128 + c) * 32 + o];
    WdT[d][o][c] = (_Float16)W_dst[((size_t)d * 128 + c) * 32 + o];
  }
  for (int t = threadIdx.x; t < 3 * 32 * 32; t += blockDim.x) {
    int d = t >> 10, rem = t & 1023;
    int o = rem >> 5, h = rem & 31;
    WvT[d][o][h] = (_Float16)W_val[((size_t)d * 32 + h) * 32 + o];
  }
  __syncthreads();

  const int lane  = threadIdx.x & 31;
  const int wid   = threadIdx.x >> 5;
  const int gwave = blockIdx.x * 4 + wid;
  const int nwaves = gridDim.x * 4;
  const int ntiles = (E + 15) >> 4;
  const int n     = lane & 15;
  const int rbase = (lane & 16) ? 8 : 0;
  const int koff  = (lane & 16) ? 16 : 0;

  for (int tile = gwave; tile < ntiles; tile += nwaves) {
    const int e0 = tile << 4;
    if (lane < 16) {
      int e = e0 + lane;
      int ec = e < E ? e : E - 1;
      s_eidx[wid][lane] = edge_index[ec];
      s_didx[wid][lane] = edge_index[E + ec];
      s_rad[wid][lane][0] = rad[(size_t)ec * 3 + 0];
      s_rad[wid][lane][1] = rad[(size_t)ec * 3 + 1];
      s_rad[wid][lane][2] = rad[(size_t)ec * 3 + 2];
    }
    wait_ds();
    // stage per-edge per-head attention weights (recompute from logits/max/denom)
#pragma unroll
    for (int q = 0; q < 4; ++q) {
      int idx = lane * 4 + q;
      int el = idx >> 3, h = idx & 7;
      int e = e0 + el;
      int ec = e < E ? e : E - 1;
      int dd = s_didx[wid][el];
      float mx = ord_dec(nodemax[(size_t)dd * 8 + h]);
      float dn = denom[(size_t)dd * 8 + h];
      s_alpha[wid][el][h] = __expf(logits[(size_t)ec * 8 + h] - mx) / (dn + 1e-9f);
    }
    wait_ds();

    const float* srow0 = node_emb + (size_t)s_eidx[wid][n] * (9 * 128);
    const float* drow0 = node_emb + (size_t)s_didx[wid][n] * (9 * 128);
    float radr[3][8];
    float alr[2][8];
    int   dsti[8];
#pragma unroll
    for (int r = 0; r < 8; ++r) {
      int m = rbase + r;
      radr[0][r] = s_rad[wid][m][0];
      radr[1][r] = s_rad[wid][m][1];
      radr[2][r] = s_rad[wid][m][2];
      alr[0][r] = s_alpha[wid][m][n >> 2];          // col n      -> head n/4
      alr[1][r] = s_alpha[wid][m][(16 + n) >> 2];   // col 16+n   -> head
      dsti[r] = s_didx[wid][m];
    }

    for (int l = 0; l < 9; ++l) {
      const int d = (l == 0) ? 0 : (l < 4 ? 1 : 2);   // DEG
      const float* srow = srow0 + l * 128;
      const float* drow = drow0 + l * 128;
      v8f acc0 = {}, acc1 = {};
#pragma unroll
      for (int kt = 0; kt < 4; ++kt) {
        v16h aS = load_a_global(srow, kt, lane);
        acc0 = wmma_f16(aS, load_b_lds(&WsT[d][n][kt * 32 + koff]), acc0);
        acc1 = wmma_f16(aS, load_b_lds(&WsT[d][16 + n][kt * 32 + koff]), acc1);
        v16h aD = load_a_global(drow, kt, lane);
        acc0 = wmma_f16(aD, load_b_lds(&WdT[d][n][kt * 32 + koff]), acc0);
        acc1 = wmma_f16(aD, load_b_lds(&WdT[d][16 + n][kt * 32 + koff]), acc1);
      }
      // rad scale + SiLU, transpose through LDS into A-layout
#pragma unroll
      for (int r = 0; r < 8; ++r) {
        float sc = radr[d][r];
        s_ms[wid][rbase + r][n]      = (_Float16)silu_f(acc0[r] * sc);
        s_ms[wid][rbase + r][16 + n] = (_Float16)silu_f(acc1[r] * sc);
      }
      wait_ds();
      v16h aM = load_a_lds32(&s_ms[wid][n][0], lane);
      v8f v0 = {}, v1 = {};
      v0 = wmma_f16(aM, load_b_lds(&WvT[d][n][koff]), v0);
      v1 = wmma_f16(aM, load_b_lds(&WvT[d][16 + n][koff]), v1);
#pragma unroll
      for (int r = 0; r < 8; ++r) {
        int e = e0 + rbase + r;
        if (e < E) {
          size_t baseo = ((size_t)dsti[r] * 9 + l) * 32;
          atomicAdd(&node_out[baseo + n],      v0[r] * alr[0][r]);
          atomicAdd(&node_out[baseo + 16 + n], v1[r] * alr[1][r]);
        }
      }
      wait_ds();   // s_ms WAR before next l
    }
  }
}

// ---------------- kernel 5: output projection + batch reduction ----------------
__global__ void __launch_bounds__(256)
k_out(const float* node_out, const float* W_out, const int* batch,
      float* out, int N, int Bn) {
  __shared__ float s_wout[3][32];
  __shared__ float s_acc[16 * 6];
  for (int t = threadIdx.x; t < 96; t += blockDim.x) s_wout[t / 32][t % 32] = W_out[t];
  for (int t = threadIdx.x; t < 16 * 6; t += blockDim.x) s_acc[t] = 0.0f;
  __syncthreads();
  int i = blockIdx.x * blockDim.x + threadIdx.x;
  int stride = gridDim.x * blockDim.x;
  for (int nd = i; nd < N; nd += stride) {
    int b = batch[nd];
    if (b < 0 || b >= 16) continue;
#pragma unroll
    for (int l = 1; l <= 6; ++l) {
      int d = (l < 4) ? 1 : 2;
      const float* row = node_out + ((size_t)nd * 9 + l) * 32;
      float s = 0.0f;
#pragma unroll
      for (int v = 0; v < 32; ++v) s += row[v] * s_wout[d][v];
      atomicAdd(&s_acc[b * 6 + (l - 1)], s);
    }
  }
  __syncthreads();
  for (int t = threadIdx.x; t < Bn * 6 && t < 96; t += blockDim.x) atomicAdd(&out[t], s_acc[t]);
}

extern "C" void kernel_launch(void* const* d_in, const int* in_sizes, int n_in,
                              void* d_out, int out_size, void* d_ws, size_t ws_size,
                              hipStream_t stream) {
  const float* node_emb      = (const float*)d_in[0];
  const int*   atomic_numbers= (const int*)d_in[1];
  const float* edge_distance = (const float*)d_in[2];
  const int*   edge_index    = (const int*)d_in[3];
  const int*   batch         = (const int*)d_in[4];
  const float* elem_emb      = (const float*)d_in[5];
  const float* W_rbf         = (const float*)d_in[6];
  const float* W_rad         = (const float*)d_in[7];
  const float* W_src         = (const float*)d_in[8];
  const float* W_dst         = (const float*)d_in[9];
  const float* W_alpha       = (const float*)d_in[10];
  const float* alpha_vec     = (const float*)d_in[11];
  const float* W_val         = (const float*)d_in[12];
  const float* W_out         = (const float*)d_in[13];

  const int N = in_sizes[1];
  const int E = in_sizes[2];
  const int maxz = in_sizes[5] / 32;
  const int Bn = out_size / 6;

  // workspace layout (floats): rad[E*3] | logits[E*8] | nodemax[N*8 u32] | denom[N*8] | node_out[N*9*32]
  float* ws       = (float*)d_ws;
  float* rad      = ws;
  float* logits   = rad + (size_t)E * 3;
  unsigned* nodemax = (unsigned*)(logits + (size_t)E * 8);
  float* denom    = (float*)(nodemax + (size_t)N * 8);
  float* node_out = denom + (size_t)N * 8;
  float* outf     = (float*)d_out;

  const long long total_no = (long long)N * 9 * 32;
  k_init<<<2048, 256, 0, stream>>>(nodemax, denom, node_out, outf, N, total_no, out_size);
  k_rad<<<1024, 256, 0, stream>>>(edge_distance, atomic_numbers, edge_index, elem_emb,
                                  W_rbf, W_rad, rad, E, maxz);
  const int ntiles = (E + 15) / 16;
  const int g1 = (ntiles + 7) / 8;     // 8 waves per 256-thread block
  k_logits<<<g1, 256, 0, stream>>>(node_emb, edge_index, W_src, W_dst, W_alpha,
                                   alpha_vec, rad, logits, nodemax, E);
  k_denom<<<2048, 256, 0, stream>>>(logits, edge_index, nodemax, denom, E);
  const int g3 = (ntiles + 3) / 4;     // 4 waves per 128-thread block
  k_msgval<<<g3, 128, 0, stream>>>(node_emb, edge_index, W_src, W_dst, W_val,
                                   rad, logits, nodemax, denom, node_out, E);
  k_out<<<1024, 256, 0, stream>>>(node_out, W_out, batch, outf, N, Bn);
}
